// RNN_16939351016109
// MI455X (gfx1250) — compile-verified, matrix operand order checked by reading, not run on hardware
//
#include <hip/hip_runtime.h>

// Problem constants (match reference)
#define Tt 1500
#define Bb 16
#define Ff 256
#define Hh 512
#define H2 1024
#define R2 32            // 2*B rows (fwd 0..15, bwd 16..31)
#define MT (R2 * Tt)     // 48000 GEMM rows

typedef __attribute__((ext_vector_type(16))) __bf16 v16bf;
typedef __attribute__((ext_vector_type(8)))  float  v8f;
typedef int vsi4 __attribute__((vector_size(16)));   // matches builtin param type

#if defined(__gfx1250__) && __has_builtin(__builtin_amdgcn_global_load_async_to_lds_b128)
#define USE_ASYNC_LDS 1
#else
#define USE_ASYNC_LDS 0
#endif

#if USE_ASYNC_LDS
__device__ __forceinline__ void async_cp_b128(const void* g, void* l) {
  __builtin_amdgcn_global_load_async_to_lds_b128(
      (__attribute__((address_space(1))) vsi4*)g,
      (__attribute__((address_space(3))) vsi4*)l, 0, 0);
}
__device__ __forceinline__ void wait_async0() {
#if __has_builtin(__builtin_amdgcn_s_wait_asynccnt)
  __builtin_amdgcn_s_wait_asynccnt(0);
#else
  asm volatile("s_wait_asynccnt 0" ::: "memory");
#endif
}
#endif

__device__ __forceinline__ unsigned short f2bf(float f) {
  unsigned int u = __float_as_uint(f);
  u += 0x7FFFu + ((u >> 16) & 1u);          // round-to-nearest-even
  return (unsigned short)(u >> 16);
}

// ---------------------------------------------------------------- converters
__global__ __launch_bounds__(256) void f32_to_bf16_kernel(
    const float* __restrict__ src, unsigned short* __restrict__ dst, int n) {
  int i = blockIdx.x * 256 + threadIdx.x;
  if (i < n) dst[i] = f2bf(src[i]);
}

// X0[r,t,f] = x[r&15, r<16 ? t : T-1-t, f]  (bidirectional concat, bf16)
__global__ __launch_bounds__(256) void build_x0_kernel(
    const float* __restrict__ x, unsigned short* __restrict__ X0) {
  int i = blockIdx.x * 256 + threadIdx.x;
  if (i >= R2 * Tt * Ff) return;
  int f = i % Ff;
  int t = (i / Ff) % Tt;
  int r = i / (Ff * Tt);
  int b = r & 15;
  int ts = (r < Bb) ? t : (Tt - 1 - t);
  X0[i] = f2bf(x[((size_t)b * Tt + ts) * Ff + f]);
}

// X1[r,t,c] = Y0[r&15, r<16 ? t : T-1-t, c]
__global__ __launch_bounds__(256) void build_x1_kernel(
    const float* __restrict__ Y0, unsigned short* __restrict__ X1) {
  int i = blockIdx.x * 256 + threadIdx.x;
  if (i >= R2 * Tt * H2) return;
  int c = i % H2;
  int t = (i / H2) % Tt;
  int r = i / (H2 * Tt);
  int b = r & 15;
  int ts = (r < Bb) ? t : (Tt - 1 - t);
  X1[i] = f2bf(Y0[((size_t)b * Tt + ts) * H2 + c]);
}

// ------------------------------------------------- input GEMM + BatchNorm
// Wout[m,n] = gamma[n]*rsqrt(var[n]+eps)*(sum_k X[m,k]*Wt[n,k] - mu[n]) + beta[n]
// block tile 128(M) x 64(N), 8 waves, k-chunks of 32.
// Staging: async global->LDS copies, double-buffered (one barrier per chunk).
#define GBM 128
#define GBN 64
#define LDA 40          // padded k-stride (ushorts): 80B keeps 16B alignment

__global__ __launch_bounds__(256) void gemm_bn_kernel(
    const unsigned short* __restrict__ X,    // [MT, K] bf16
    const unsigned short* __restrict__ Wt,   // [1024, K] bf16 (Ww)
    const float* __restrict__ gam, const float* __restrict__ bet,
    const float* __restrict__ mu,  const float* __restrict__ var,
    float* __restrict__ Wout, int K) {
  __shared__ unsigned short aSh[2][GBM * LDA];
  __shared__ unsigned short bSh[2][GBN * LDA];
  const int tid  = threadIdx.x;
  const int lane = tid & 31;
  const int wave = tid >> 5;
  const int ml   = lane & 15;
  const int hi   = lane >> 4;
  const int bm   = blockIdx.x * GBM;
  const int bn   = blockIdx.y * GBN;

  // staging roles (fixed per thread)
  const int ar  = tid >> 1;          // A row 0..127
  const int ako = (tid & 1) * 16;    // A k-offset 0/16
  const int br  = tid >> 2;          // B n-row 0..63
  const int bko = (tid & 3) * 8;     // B k-offset
  const unsigned short* aPtr = X  + (size_t)(bm + ar) * K + ako;
  const unsigned short* bPtr = Wt + (size_t)(bn + br) * K + bko;

  v8f acc[4] = {};

#if USE_ASYNC_LDS
  // prologue: chunk 0 -> buffer 0
  async_cp_b128(aPtr,     &aSh[0][ar * LDA + ako]);
  async_cp_b128(aPtr + 8, &aSh[0][ar * LDA + ako + 8]);
  async_cp_b128(bPtr,     &bSh[0][br * LDA + bko]);
  wait_async0();
  __syncthreads();

  int cur = 0;
  for (int kk = 0; kk < K; kk += 32) {
    if (kk + 32 < K) {   // prefetch next chunk into alternate buffer
      async_cp_b128(aPtr + kk + 32,     &aSh[cur ^ 1][ar * LDA + ako]);
      async_cp_b128(aPtr + kk + 32 + 8, &aSh[cur ^ 1][ar * LDA + ako + 8]);
      async_cp_b128(bPtr + kk + 32,     &bSh[cur ^ 1][br * LDA + bko]);
    }
    const unsigned short* aS = aSh[cur];
    const unsigned short* bS = bSh[cur];

    v16bf af;
#pragma unroll
    for (int vi = 0; vi < 8; ++vi) {
      int kb = (vi < 4 ? 0 : 16) + hi * 8 + (vi & 3) * 2;
      ((unsigned int*)&af)[vi] =
          *(const unsigned int*)&aS[(wave * 16 + ml) * LDA + kb];
    }
#pragma unroll
    for (int nt = 0; nt < 4; ++nt) {
      v16bf bfr;
#pragma unroll
      for (int vi = 0; vi < 8; ++vi) {
        int kb = (vi < 4 ? 0 : 16) + hi * 8 + (vi & 3) * 2;
        ((unsigned int*)&bfr)[vi] =
            *(const unsigned int*)&bS[(nt * 16 + ml) * LDA + kb];
      }
      acc[nt] = __builtin_amdgcn_wmma_f32_16x16x32_bf16(
          false, af, false, bfr, (short)0, acc[nt], false, false);
    }
    wait_async0();       // next chunk has landed in LDS
    __syncthreads();     // everyone done reading current buffer
    cur ^= 1;
  }
#else
  for (int kk = 0; kk < K; kk += 32) {
    {   // stage A: two uint4 per thread
      const unsigned short* src = aPtr + kk;
      *(uint4*)&aSh[0][ar * LDA + ako]     = *(const uint4*)src;
      *(uint4*)&aSh[0][ar * LDA + ako + 8] = *(const uint4*)(src + 8);
    }
    {   // stage B: one uint4 per thread
      *(uint4*)&bSh[0][br * LDA + bko] = *(const uint4*)(bPtr + kk);
    }
    __syncthreads();

    v16bf af;
#pragma unroll
    for (int vi = 0; vi < 8; ++vi) {
      int kb = (vi < 4 ? 0 : 16) + hi * 8 + (vi & 3) * 2;
      ((unsigned int*)&af)[vi] =
          *(const unsigned int*)&aSh[0][(wave * 16 + ml) * LDA + kb];
    }
#pragma unroll
    for (int nt = 0; nt < 4; ++nt) {
      v16bf bfr;
#pragma unroll
      for (int vi = 0; vi < 8; ++vi) {
        int kb = (vi < 4 ? 0 : 16) + hi * 8 + (vi & 3) * 2;
        ((unsigned int*)&bfr)[vi] =
            *(const unsigned int*)&bSh[0][(nt * 16 + ml) * LDA + kb];
      }
      acc[nt] = __builtin_amdgcn_wmma_f32_16x16x32_bf16(
          false, af, false, bfr, (short)0, acc[nt], false, false);
    }
    __syncthreads();
  }
#endif

#pragma unroll
  for (int nt = 0; nt < 4; ++nt) {
    int n = bn + nt * 16 + ml;
    float s  = gam[n] * rsqrtf(var[n] + 1e-5f);
    float sh = bet[n] - mu[n] * s;
#pragma unroll
    for (int vi = 0; vi < 8; ++vi) {
      int mrow = bm + wave * 16 + vi + hi * 8;   // C layout: VGPR i -> M = i + 8*hi
      Wout[(size_t)mrow * H2 + n] = acc[nt][vi] * s + sh;
    }
  }
}

// ---------------------------------------------------------- liGRU recurrence
// One block per direction (blockIdx.x: 0 fwd rows 0..15 of Wg, 1 bwd 16..31).
// 32 waves; wave w owns output columns [32w, 32w+32) (2 WMMA N-tiles).
// h kept as bf16 in LDS (A matrix) + fp32 in the owning thread's registers.
// Wu streamed from L2 every step (1MB, L2-resident).
#define HBPAD 516       // padded row stride of h_bf (ushorts)

__global__ __launch_bounds__(1024) void ligru_rec_kernel(
    const float* __restrict__ Wg,           // [32, T, 1024] BN'd gate inputs
    const unsigned short* __restrict__ Wu,  // [1024, 512] bf16
    float* __restrict__ Y) {                // [16, T, 1024] layer output
  extern __shared__ char smem[];
  float* gates = (float*)smem;                                  // 16*1024 f32
  unsigned short* hbf = (unsigned short*)(smem + Bb * H2 * 4);  // 16*HBPAD

  const int tid  = threadIdx.x;
  const int lane = tid & 31;
  const int wave = tid >> 5;       // 0..31
  const int ml   = lane & 15;
  const int hi   = lane >> 4;
  const int dir  = blockIdx.x;

  const float* wg = Wg + (size_t)dir * Bb * Tt * H2;

  // update-phase roles: thread owns fixed (row, 8 cols) of h forever
  const int ur = tid >> 6;          // 0..15
  const int uc = (tid & 63) * 8;    // 0..504

  float hreg[8];
#pragma unroll
  for (int j = 0; j < 8; ++j) hreg[j] = 0.0f;
  for (int i = tid; i < Bb * Hh; i += 1024)
    hbf[(i >> 9) * HBPAD + (i & 511)] = 0;
  __syncthreads();

  const int n0 = wave * 32 + ml;
  const int n1 = n0 + 16;

  for (int t = 0; t < Tt; ++t) {
    // --- gates[0:16, n0/n1] = h @ Wu^T  (16 K-tiles of 32)
    v8f acc0 = {}, acc1 = {};
#pragma unroll
    for (int kt = 0; kt < 16; ++kt) {
      v16bf af, bf0, bf1;
#pragma unroll
      for (int vi = 0; vi < 8; ++vi) {
        int kb = kt * 32 + (vi < 4 ? 0 : 16) + hi * 8 + (vi & 3) * 2;
        ((unsigned int*)&af)[vi]  = *(const unsigned int*)&hbf[ml * HBPAD + kb];
        ((unsigned int*)&bf0)[vi] = *(const unsigned int*)&Wu[(size_t)n0 * Hh + kb];
        ((unsigned int*)&bf1)[vi] = *(const unsigned int*)&Wu[(size_t)n1 * Hh + kb];
      }
      acc0 = __builtin_amdgcn_wmma_f32_16x16x32_bf16(
          false, af, false, bf0, (short)0, acc0, false, false);
      acc1 = __builtin_amdgcn_wmma_f32_16x16x32_bf16(
          false, af, false, bf1, (short)0, acc1, false, false);
    }
#pragma unroll
    for (int vi = 0; vi < 8; ++vi) {
      int mrow = vi + hi * 8;
      gates[mrow * H2 + n0] = acc0[vi];
      gates[mrow * H2 + n1] = acc1[vi];
    }
    __syncthreads();

    // --- gating update: h = z*h + (1-z)*relu(a)
    const float* wrow = wg + ((size_t)ur * Tt + t) * H2;
    const int tt = dir ? (Tt - 1 - t) : t;
    float* yrow = Y + ((size_t)ur * Tt + tt) * H2 + dir * Hh;
#pragma unroll
    for (int j = 0; j < 8; ++j) {
      int c = uc + j;
      float a  = gates[ur * H2 + c]      + wrow[c];
      float zt = gates[ur * H2 + Hh + c] + wrow[Hh + c];
      float z  = 1.0f / (1.0f + __expf(-zt));
      float hc = fmaxf(a, 0.0f);
      float h  = z * hreg[j] + (1.0f - z) * hc;
      hreg[j] = h;
      hbf[ur * HBPAD + c] = f2bf(h);
      yrow[c] = h;
    }
    __syncthreads();
  }
}

// --------------------------------------------------------------------- host
extern "C" void kernel_launch(void* const* d_in, const int* in_sizes, int n_in,
                              void* d_out, int out_size, void* d_ws, size_t ws_size,
                              hipStream_t stream) {
  const float* x   = (const float*)d_in[0];
  const float* Ww0 = (const float*)d_in[1];
  const float* Wu0 = (const float*)d_in[2];
  const float* g0  = (const float*)d_in[3];
  const float* b0  = (const float*)d_in[4];
  const float* m0  = (const float*)d_in[5];
  const float* v0  = (const float*)d_in[6];
  const float* Ww1 = (const float*)d_in[7];
  const float* Wu1 = (const float*)d_in[8];
  const float* g1  = (const float*)d_in[9];
  const float* b1  = (const float*)d_in[10];
  const float* m1  = (const float*)d_in[11];
  const float* v1  = (const float*)d_in[12];
  float* out = (float*)d_out;

  char* ws = (char*)d_ws;
  size_t off = 0;
  unsigned short* X0 = (unsigned short*)(ws + off); off += (size_t)R2 * Tt * Ff * 2;   // 24.6 MB
  unsigned short* X1 = (unsigned short*)(ws + off); off += (size_t)R2 * Tt * H2 * 2;   // 98.3 MB
  float* Wg          = (float*)(ws + off);          off += (size_t)R2 * Tt * H2 * 4;   // 196.6 MB
  float* Y0          = (float*)(ws + off);          off += (size_t)Bb * Tt * H2 * 4;   // 98.3 MB
  unsigned short* Wu0b = (unsigned short*)(ws + off); off += (size_t)H2 * Hh * 2;      // 1 MB
  unsigned short* Wu1b = (unsigned short*)(ws + off); off += (size_t)H2 * Hh * 2;      // 1 MB
  unsigned short* Ww0b = (unsigned short*)(ws + off); off += (size_t)H2 * Ff * 2;      // 0.5 MB
  unsigned short* Ww1b = (unsigned short*)(ws + off); off += (size_t)H2 * H2 * 2;      // 2 MB

  const int recSmem = Bb * H2 * 4 + Bb * HBPAD * 2;  // 82048 bytes
  (void)hipFuncSetAttribute((const void*)ligru_rec_kernel,
                            hipFuncAttributeMaxDynamicSharedMemorySize, recSmem);

  // weight / input conversions to bf16
  f32_to_bf16_kernel<<<(H2 * Hh + 255) / 256, 256, 0, stream>>>(Wu0, Wu0b, H2 * Hh);
  f32_to_bf16_kernel<<<(H2 * Hh + 255) / 256, 256, 0, stream>>>(Wu1, Wu1b, H2 * Hh);
  f32_to_bf16_kernel<<<(H2 * Ff + 255) / 256, 256, 0, stream>>>(Ww0, Ww0b, H2 * Ff);
  f32_to_bf16_kernel<<<(H2 * H2 + 255) / 256, 256, 0, stream>>>(Ww1, Ww1b, H2 * H2);
  build_x0_kernel<<<(R2 * Tt * Ff + 255) / 256, 256, 0, stream>>>(x, X0);

  // ---- layer 0
  gemm_bn_kernel<<<dim3(MT / GBM, H2 / GBN), 256, 0, stream>>>(
      X0, Ww0b, g0, b0, m0, v0, Wg, Ff);
  ligru_rec_kernel<<<2, 1024, recSmem, stream>>>(Wg, Wu0b, Y0);

  // ---- layer 1
  build_x1_kernel<<<(R2 * Tt * H2 + 255) / 256, 256, 0, stream>>>(Y0, X1);
  gemm_bn_kernel<<<dim3(MT / GBM, H2 / GBN), 256, 0, stream>>>(
      X1, Ww1b, g1, b1, m1, v1, Wg, H2);
  ligru_rec_kernel<<<2, 1024, recSmem, stream>>>(Wg, Wu1b, out);
}
